// SelfAttention_23330262352109
// MI455X (gfx1250) — matrix operand order repaired and checked by reading, not compiled, not run on hardware
//
#include <hip/hip_runtime.h>

typedef unsigned short u16;
typedef __attribute__((ext_vector_type(16))) _Float16 v16h;
typedef __attribute__((ext_vector_type(8)))  _Float16 v8h;
typedef __attribute__((ext_vector_type(8)))  float    v8f;
typedef __attribute__((ext_vector_type(4)))  unsigned int u32x4;
typedef __attribute__((ext_vector_type(8)))  unsigned int u32x8;

#define BB 16
#define TT 768
#define CC 1024
#define HH 16
#define DD 64
#define KDIM 1024

__device__ __forceinline__ u16 f2h(float f) {
  return __builtin_bit_cast(u16, (_Float16)f);
}
__device__ __forceinline__ float h2f(u16 h) {
  return (float)__builtin_bit_cast(_Float16, h);
}
__device__ __forceinline__ v16h cat8(v8h a, v8h b) {
  return __builtin_shufflevector(a, b, 0,1,2,3,4,5,6,7,8,9,10,11,12,13,14,15);
}
__device__ __forceinline__ v8f f16mma(v16h a, v16h b, v8f c) {
  return __builtin_amdgcn_wmma_f32_16x16x32_f16(false, a, false, b, (short)0, c,
                                                false, false);
}

// ---------------------------------------------------------------- TDM load
// Post a 2D tile load: 128 rows x 32 f16 (64B) per row, row stride KDIM
// elements, into LDS at lds_off. D# layout per CDNA5 ISA ch.8 (group0 128b,
// group1 256b; groups 2/3 NULL for 2D tensors).
__device__ __forceinline__ void tdm_load_tile(const u16* g, unsigned lds_off) {
  unsigned long long ga = (unsigned long long)(uintptr_t)g;
  u32x4 g0 = { 1u,                                   // count=1, user mode
               lds_off,                              // lds_addr [63:32]
               (unsigned)ga,                         // global_addr lo
               (unsigned)(ga >> 32) | (2u << 30) };  // global_addr hi | type=2
  u32x8 g1 = { (1u << 16),          // workgroup_mask=0 | data_size=1 (2 bytes)
               (KDIM & 0xffffu) << 16,   // tensor_dim0 lo16 = 1024
               (128u << 16),             // tensor_dim0 hi=0 | tensor_dim1 lo16=128
               (32u << 16),              // tensor_dim1 hi=0 | tile_dim0=32
               128u,                     // tile_dim1=128 | tile_dim2=0
               (unsigned)KDIM,           // tensor_dim0_stride lo32 = 1024
               0u,                       // stride hi | dim1_stride lo
               0u };
  asm volatile("tensor_load_to_lds %0, %1" :: "s"(g0), "s"(g1) : "memory");
}

// ---------------------------------------------------------------- convert
__global__ void cvt_kernel(const float* __restrict__ src, u16* __restrict__ dst,
                           int n) {
  int i = blockIdx.x * blockDim.x + threadIdx.x;
  if (i < n) dst[i] = f2h(src[i]);
}

// ---------------------------------------------------------------- GEMM core
// A: [M][K] f16 row-major, W: [N][K] f16 row-major; computes C += A * W^T
// 128x128 block tile, 8 waves as 2(m) x 4(n), each wave 4x2 16x16 tiles.
// Tiles are streamed by the Tensor Data Mover into double-buffered LDS.
__device__ __forceinline__ void gemm_tile(const u16* __restrict__ Ag,
                                          const u16* __restrict__ Wg,
                                          u16 (*As)[128 * 32],
                                          u16 (*Bs)[128 * 32],
                                          v8f acc[4][2]) {
  const int tid = threadIdx.x;
  const int lane = tid & 31, l16 = lane & 15, hi = lane >> 4;
  const int wave = tid >> 5;
  const int wm = wave >> 2, wn = wave & 3;
  const unsigned asb0 = (unsigned)(uintptr_t)&As[0][0];
  const unsigned asb1 = (unsigned)(uintptr_t)&As[1][0];
  const unsigned bsb0 = (unsigned)(uintptr_t)&Bs[0][0];
  const unsigned bsb1 = (unsigned)(uintptr_t)&Bs[1][0];
  if (wave == 0) {                       // prologue: fill buffer 0
    tdm_load_tile(Ag, asb0);
    tdm_load_tile(Wg, bsb0);
  }
  __builtin_amdgcn_s_wait_tensorcnt(0);
  __syncthreads();
  for (int kb = 0; kb < KDIM; kb += 32) {
    const int cur = (kb >> 5) & 1;
    if ((kb + 32 < KDIM) && (wave == 0)) {   // prefetch next tile via TDM
      tdm_load_tile(Ag + kb + 32, cur ? asb0 : asb1);
      tdm_load_tile(Wg + kb + 32, cur ? bsb0 : bsb1);
    }
    const u16* Asc = As[cur];
    const u16* Bsc = Bs[cur];
    v16h bfr[2];
#pragma unroll
    for (int nt = 0; nt < 2; ++nt) {
      const u16* p = &Bsc[(wn * 32 + nt * 16 + l16) * 32 + hi * 8];
      bfr[nt] = cat8(*(const v8h*)p, *(const v8h*)(p + 16)); // K = hi*8+{0..7,16..23}
    }
#pragma unroll
    for (int mt = 0; mt < 4; ++mt) {
      const u16* p = &Asc[(wm * 64 + mt * 16 + l16) * 32 + hi * 8];
      v16h afr = cat8(*(const v8h*)p, *(const v8h*)(p + 16)); // K = hi*8+{0..7,16..23}
#pragma unroll
      for (int nt = 0; nt < 2; ++nt)
        acc[mt][nt] = f16mma(afr, bfr[nt], acc[mt][nt]);
    }
    __builtin_amdgcn_s_wait_tensorcnt(0);  // no-op for waves 1..7
    __syncthreads();
  }
}

// ---------------------------------------------------------------- QKV GEMM
// out n in [0,3072): which=n>>10 (q/k/v); q,k -> [B*H][T][D], v -> [B*H][D][T]
__global__ __launch_bounds__(256) void qkv_gemm_kernel(
    const u16* __restrict__ xh, const u16* __restrict__ wqh,
    const float* __restrict__ bias, u16* __restrict__ qb, u16* __restrict__ kb_,
    u16* __restrict__ vtb) {
  __shared__ alignas(32) u16 As[2][128 * 32];
  __shared__ alignas(32) u16 Bs[2][128 * 32];
  v8f acc[4][2] = {};
  const u16* Ag = xh + (size_t)blockIdx.x * 128 * KDIM;
  const u16* Wg = wqh + (size_t)blockIdx.y * 128 * KDIM;
  gemm_tile(Ag, Wg, As, Bs, acc);
  const int lane = threadIdx.x & 31, l16 = lane & 15, hi = lane >> 4;
  const int wave = threadIdx.x >> 5, wm = wave >> 2, wn = wave & 3;
#pragma unroll
  for (int nt = 0; nt < 2; ++nt) {
    int n = blockIdx.y * 128 + wn * 32 + nt * 16 + l16;
    float bv = bias[n];
    int which = n >> 10;
    int c = n & (CC - 1);
    int h = c >> 6, d = c & 63;
#pragma unroll
    for (int mt = 0; mt < 4; ++mt) {
#pragma unroll
      for (int r = 0; r < 8; ++r) {
        int m = blockIdx.x * 128 + wm * 64 + mt * 16 + r + hi * 8;
        int b = m / TT, t = m - b * TT;
        u16 o = f2h(acc[mt][nt][r] + bv);
        size_t bh = (size_t)(b * HH + h);
        if (which == 0)      qb [(bh * TT + t) * DD + d] = o;
        else if (which == 1) kb_[(bh * TT + t) * DD + d] = o;
        else                 vtb[(bh * DD + d) * TT + t] = o;   // transposed V
      }
    }
  }
}

// ---------------------------------------------------------------- RoPE
// In-place on q (fused 1/sqrt(D) scale) and k. Pair channels over full C.
__global__ void rope_kernel(u16* __restrict__ q, u16* __restrict__ k) {
  int i = blockIdx.x * blockDim.x + threadIdx.x;
  const int total = BB * HH * TT * (DD / 2);
  if (i >= total) return;
  int d2 = i & 31;
  int it = i >> 5;            // bh*TT + t
  int bh = it / TT;
  int t = it - bh * TT;
  int h = bh & (HH - 1);
  int j2 = h * DD + 2 * d2;   // = 2*j  (pair index over full C)
  float inv_freq = exp2f(-(float)j2 * (13.2877123795494f / 1024.0f)); // 10000^(-j2/C)
  float ang = (float)t * inv_freq;
  float sn, cs;
  sincosf(ang, &sn, &cs);
  size_t base = ((size_t)bh * TT + t) * DD + 2 * d2;
  float ur = h2f(q[base]), ui = h2f(q[base + 1]);
  q[base]     = f2h((ur * cs - ui * sn) * 0.125f);
  q[base + 1] = f2h((ur * sn + ui * cs) * 0.125f);
  ur = h2f(k[base]); ui = h2f(k[base + 1]);
  k[base]     = f2h(ur * cs - ui * sn);
  k[base + 1] = f2h(ur * sn + ui * cs);
}

// ---------------------------------------------------------------- attention
// grid: (B*H, T/128), 8 waves; wave owns 16 q rows, barrier-free flash loop.
__global__ __launch_bounds__(256) void attn_kernel(
    const u16* __restrict__ qg, const u16* __restrict__ kg,
    const u16* __restrict__ vtg, u16* __restrict__ yg) {
  const int bh = blockIdx.x;
  const int h = bh & (HH - 1);
  const int b = bh >> 4;
  const int wave = threadIdx.x >> 5;
  const int lane = threadIdx.x & 31;
  const int l16 = lane & 15, hi = lane >> 4;
  const int qr0 = blockIdx.y * 128 + wave * 16;
  const u16* qp = qg + (size_t)bh * TT * DD;
  const u16* kp = kg + (size_t)bh * TT * DD;
  const u16* vp = vtg + (size_t)bh * DD * TT;
  __shared__ alignas(32) u16 pst[8][16 * 64];   // per-wave P staging
  u16* myp = pst[wave];

  v16h qa[2];
  {
    const u16* base = qp + (size_t)(qr0 + l16) * DD;
#pragma unroll
    for (int ks = 0; ks < 2; ++ks) {
      const u16* p = base + ks * 32 + hi * 8;
      qa[ks] = cat8(*(const v8h*)p, *(const v8h*)(p + 16));
    }
  }
  v8f acc[4] = {};
  float mrow[8], lrow[8];
#pragma unroll
  for (int r = 0; r < 8; ++r) { mrow[r] = -1e30f; lrow[r] = 0.f; }

  const float LOG2E = 1.4426950408889634f;
  const int kend = qr0 + 16;                      // causal: keys <= last q row
  for (int kb = 0; kb < kend; kb += 64) {
    v8f s[4];
#pragma unroll
    for (int nt = 0; nt < 4; ++nt) {
      v8f sa = {};
      const u16* krow = kp + (size_t)(kb + nt * 16 + l16) * DD + hi * 8;
#pragma unroll
      for (int ks = 0; ks < 2; ++ks)
        sa = f16mma(qa[ks], cat8(*(const v8h*)(krow + ks * 32), *(const v8h*)(krow + ks * 32 + 16)), sa);
      s[nt] = sa;
    }
    // causal mask
#pragma unroll
    for (int nt = 0; nt < 4; ++nt) {
      int key = kb + nt * 16 + l16;
#pragma unroll
      for (int r = 0; r < 8; ++r)
        if (key > qr0 + r + hi * 8) s[nt][r] = -1e30f;
    }
    // row max (rows per-VGPR, 16 columns per-lane within each half)
    float bm[8];
#pragma unroll
    for (int r = 0; r < 8; ++r)
      bm[r] = fmaxf(fmaxf(s[0][r], s[1][r]), fmaxf(s[2][r], s[3][r]));
#pragma unroll
    for (int off = 8; off >= 1; off >>= 1)
#pragma unroll
      for (int r = 0; r < 8; ++r)
        bm[r] = fmaxf(bm[r], __shfl_xor(bm[r], off, 16));
    float ef[8];
#pragma unroll
    for (int r = 0; r < 8; ++r) {
      float nm = fmaxf(mrow[r], bm[r]);
      ef[r] = exp2f((mrow[r] - nm) * LOG2E);
      mrow[r] = nm;
    }
#pragma unroll
    for (int dt = 0; dt < 4; ++dt)
#pragma unroll
      for (int r = 0; r < 8; ++r)
        acc[dt][r] *= ef[r];
#pragma unroll
    for (int nt = 0; nt < 4; ++nt)
#pragma unroll
      for (int r = 0; r < 8; ++r)
        s[nt][r] = exp2f((s[nt][r] - mrow[r]) * LOG2E);
    float ls[8];
#pragma unroll
    for (int r = 0; r < 8; ++r)
      ls[r] = (s[0][r] + s[1][r]) + (s[2][r] + s[3][r]);
#pragma unroll
    for (int off = 8; off >= 1; off >>= 1)
#pragma unroll
      for (int r = 0; r < 8; ++r)
        ls[r] += __shfl_xor(ls[r], off, 16);
#pragma unroll
    for (int r = 0; r < 8; ++r)
      lrow[r] = lrow[r] * ef[r] + ls[r];
    // stage P (C/D layout -> row-major [16][64]) through private LDS
#pragma unroll
    for (int nt = 0; nt < 4; ++nt)
#pragma unroll
      for (int r = 0; r < 8; ++r)
        myp[(r + hi * 8) * 64 + nt * 16 + l16] = f2h(s[nt][r]);
    __builtin_amdgcn_wave_barrier();     // same-wave LDS is in-order in HW
    v16h pa[2];
#pragma unroll
    for (int ks = 0; ks < 2; ++ks) {
      const u16* p = myp + l16 * 64 + ks * 32 + hi * 8;
      pa[ks] = cat8(*(const v8h*)p, *(const v8h*)(p + 16));
    }
#pragma unroll
    for (int dt = 0; dt < 4; ++dt) {
      const u16* vrow = vp + (size_t)(dt * 16 + l16) * TT + kb + hi * 8;
#pragma unroll
      for (int ks = 0; ks < 2; ++ks)
        acc[dt] = f16mma(pa[ks], cat8(*(const v8h*)(vrow + ks * 32), *(const v8h*)(vrow + ks * 32 + 16)), acc[dt]);
    }
    __builtin_amdgcn_wave_barrier();
  }
  float inv[8];
#pragma unroll
  for (int r = 0; r < 8; ++r) inv[r] = 1.0f / lrow[r];
#pragma unroll
  for (int dt = 0; dt < 4; ++dt) {
    int d = dt * 16 + l16;
#pragma unroll
    for (int r = 0; r < 8; ++r) {
      int t = qr0 + r + hi * 8;
      yg[((size_t)b * TT + t) * CC + h * DD + d] = f2h(acc[dt][r] * inv[r]);
    }
  }
}

// ---------------------------------------------------------------- out GEMM
__global__ __launch_bounds__(256) void out_gemm_kernel(
    const u16* __restrict__ yh, const u16* __restrict__ woh,
    const float* __restrict__ bias, float* __restrict__ out) {
  __shared__ alignas(32) u16 As[2][128 * 32];
  __shared__ alignas(32) u16 Bs[2][128 * 32];
  v8f acc[4][2] = {};
  const u16* Ag = yh + (size_t)blockIdx.x * 128 * KDIM;
  const u16* Wg = woh + (size_t)blockIdx.y * 128 * KDIM;
  gemm_tile(Ag, Wg, As, Bs, acc);
  const int lane = threadIdx.x & 31, l16 = lane & 15, hi = lane >> 4;
  const int wave = threadIdx.x >> 5, wm = wave >> 2, wn = wave & 3;
#pragma unroll
  for (int nt = 0; nt < 2; ++nt) {
    int n = blockIdx.y * 128 + wn * 32 + nt * 16 + l16;
    float bv = bias[n];
#pragma unroll
    for (int mt = 0; mt < 4; ++mt) {
#pragma unroll
      for (int r = 0; r < 8; ++r) {
        int m = blockIdx.x * 128 + wm * 64 + mt * 16 + r + hi * 8;
        out[(size_t)m * CC + n] = acc[mt][nt][r] + bv;
      }
    }
  }
}

// ---------------------------------------------------------------- launch
extern "C" void kernel_launch(void* const* d_in, const int* in_sizes, int n_in,
                              void* d_out, int out_size, void* d_ws, size_t ws_size,
                              hipStream_t stream) {
  const float* x    = (const float*)d_in[0];
  const float* Wqkv = (const float*)d_in[1];
  const float* bqkv = (const float*)d_in[2];
  const float* Wo   = (const float*)d_in[3];
  const float* bo   = (const float*)d_in[4];
  float* out = (float*)d_out;

  // workspace partition (f16/u16 elements); total = 128 MiB
  u16* xh  = (u16*)d_ws;                               // 12288x1024
  u16* wqh = xh  + (size_t)BB * TT * CC;               // 3072x1024
  u16* woh = wqh + (size_t)3 * CC * CC;                // 1024x1024
  u16* qb  = woh + (size_t)CC * CC;                    // [B*H][T][D]
  u16* kb  = qb  + (size_t)BB * TT * CC;
  u16* vtb = kb  + (size_t)BB * TT * CC;               // [B*H][D][T]
  u16* yb  = vtb + (size_t)BB * TT * CC;               // [B][T][C]

  int nx = BB * TT * CC;
  cvt_kernel<<<(nx + 255) / 256, 256, 0, stream>>>(x, xh, nx);
  int nw = 3 * CC * CC;
  cvt_kernel<<<(nw + 255) / 256, 256, 0, stream>>>(Wqkv, wqh, nw);
  int no = CC * CC;
  cvt_kernel<<<(no + 255) / 256, 256, 0, stream>>>(Wo, woh, no);

  dim3 g1((BB * TT) / 128, (3 * CC) / 128);            // 96 x 24
  qkv_gemm_kernel<<<g1, 256, 0, stream>>>(xh, wqh, bqkv, qb, kb, vtb);

  int nr = BB * HH * TT * (DD / 2);
  rope_kernel<<<(nr + 255) / 256, 256, 0, stream>>>(qb, kb);

  dim3 g2(BB * HH, TT / 128);                          // 256 x 6
  attn_kernel<<<g2, 256, 0, stream>>>(qb, kb, vtb, yb);

  dim3 g3((BB * TT) / 128, CC / 128);                  // 96 x 8
  out_gemm_kernel<<<g3, 256, 0, stream>>>(yb, woh, bo, out);
}
